// MultiBoxLoss_four_corners_54271206752708
// MI455X (gfx1250) — compile-verified, matrix operand order checked by reading, not run on hardware
//
#include <hip/hip_runtime.h>

#define NB 64
#define NP 8732
#define NC 81
#define TILE_P 96        // priors per block = 6 waves * 16 rows
#define NTHR  192
#define ECOLS 96         // padded class columns (3 x K=32)
#define SORT_N 16384
#define NEG_POS_RATIO 3

typedef __attribute__((ext_vector_type(16))) _Float16     v16h;
typedef __attribute__((ext_vector_type(8)))  _Float16     v8h;
typedef __attribute__((ext_vector_type(8)))  float        v8f;
typedef __attribute__((ext_vector_type(4)))  unsigned int v4u;
typedef __attribute__((ext_vector_type(8)))  int          v8i_t;
typedef __attribute__((ext_vector_type(4)))  int          v4i_t;

// workspace layout (32-bit slots):
//   [0] loss_l   [1] loss_fc   [2] loss_c   [4] total_pos (int)
//   [8..71] per-batch-row positive count (int)
//   [256 ...] mine[NB*NP] floats (CE of negatives, 0 for positives)

__device__ __forceinline__ float wave_sum_f(float v) {
#pragma unroll
  for (int off = 16; off > 0; off >>= 1) v += __shfl_xor(v, off, 32);
  return v;
}
__device__ __forceinline__ int wave_sum_i(int v) {
#pragma unroll
  for (int off = 16; off > 0; off >>= 1) v += __shfl_xor(v, off, 32);
  return v;
}

__global__ void zero_ws_kernel(int* iws) {
  if (threadIdx.x < 128) iws[threadIdx.x] = 0;
}

__global__ void smoothl1_kernel(const float* __restrict__ loc_d,
                                const float* __restrict__ loc_t,
                                const float* __restrict__ fc_d,
                                const float* __restrict__ fc_t,
                                const int*   __restrict__ conf_t,
                                float* fws) {
  const int idx = blockIdx.x * blockDim.x + threadIdx.x;
  float sl = 0.f, sf = 0.f;
  if (idx < NB * NP) {
    if (idx + (int)blockDim.x < NB * NP)  // gfx1250 global_prefetch_b8
      __builtin_prefetch(&fc_d[(size_t)(idx + blockDim.x) * 8], 0, 0);
    if (conf_t[idx] > 0) {
#pragma unroll
      for (int j = 0; j < 4; ++j) {
        float d = fabsf(loc_d[(size_t)idx * 4 + j] - loc_t[(size_t)idx * 4 + j]);
        sl += (d < 1.f) ? 0.5f * d * d : d - 0.5f;
      }
#pragma unroll
      for (int j = 0; j < 8; ++j) {
        float d = fabsf(fc_d[(size_t)idx * 8 + j] - fc_t[(size_t)idx * 8 + j]);
        sf += (d < 1.f) ? 0.5f * d * d : d - 0.5f;
      }
    }
  }
  sl = wave_sum_f(sl);
  sf = wave_sum_f(sf);
  if ((threadIdx.x & 31) == 0) {
    if (sl != 0.f) atomicAdd(&fws[0], sl);
    if (sf != 0.f) atomicAdd(&fws[1], sf);
  }
}

__global__ void __launch_bounds__(NTHR)
ce_kernel(const float* __restrict__ conf,
          const int*   __restrict__ conf_t,
          float* fws, int* iws,
          float* __restrict__ mine) {
  __shared__ float tile[TILE_P * NC];                    // 31104 B, TDM-staged f32
  __shared__ _Float16 __attribute__((aligned(16))) et[TILE_P * ECOLS]; // 18432 B
  __shared__ float rmax[TILE_P];
  __shared__ float wsum[(NTHR / 32) * 16];
  const int b    = blockIdx.y;
  const int p0   = blockIdx.x * TILE_P;
  const int np   = min(TILE_P, NP - p0);
  const int lane = threadIdx.x & 31;
  const int w    = threadIdx.x >> 5;
  const int hi   = lane >> 4;
  const int m    = lane & 15;
  const int r    = w * 16 + m;

  // ---- Tensor Data Mover: DMA the [np x 81] f32 tile into LDS ----
  {
    const uint64_t ga = (uint64_t)(uintptr_t)(conf + ((size_t)b * NP + p0) * NC);
    const uint32_t lo = (uint32_t)(uintptr_t)(&tile[0]);  // flat->LDS offset
    v4u g0;  v8i_t g1;  v4i_t gz4;  v8i_t gz8;
    g0.x = 1u;                                            // count=1, valid user D#
    g0.y = lo;                                            // lds_addr
    g0.z = (uint32_t)ga;                                  // global_addr[31:0]
    g0.w = (uint32_t)((ga >> 32) & 0x01FFFFFFu) | 0x80000000u; // addr[56:32]|type=2
    g1.s0 = 0x00020000;                                   // data_size=2 (4B)
    g1.s1 = (int)(81u << 16);                             // tensor_dim0 = 81
    g1.s2 = (int)(((unsigned)np) << 16);                  // tensor_dim1 = np
    g1.s3 = (int)(81u << 16);                             // tile_dim0 = 81
    g1.s4 = np;                                           // tile_dim1 = np
    g1.s5 = 81;                                           // tensor_dim0_stride
    g1.s6 = 0;
    g1.s7 = 0;
    gz4.x = gz4.y = gz4.z = gz4.w = 0;
    gz8.s0 = gz8.s1 = gz8.s2 = gz8.s3 = 0;
    gz8.s4 = gz8.s5 = gz8.s6 = gz8.s7 = 0;
    if (w == 0) {
      __builtin_amdgcn_tensor_load_to_lds(g0, g1, gz4, gz4, gz8, 0);
      __builtin_amdgcn_s_wait_tensorcnt(0);
    }
  }
  __syncthreads();

  // ---- per-prior max, classes split across the two half-waves ----
  float mx = -3.0e38f;
  if (r < np) {
    for (int c = hi; c < NC; c += 2) mx = fmaxf(mx, tile[r * NC + c]);
  }
  mx = fmaxf(mx, __shfl_xor(mx, 16, 32));
  if (hi == 0) rmax[r] = mx;
  __syncthreads();

  // ---- cooperatively materialize exp(x - rowmax) as padded f16 tile ----
  for (int i = threadIdx.x; i < TILE_P * (ECOLS / 2); i += NTHR) {
    const int rr = i / (ECOLS / 2);
    const int c0 = (i - rr * (ECOLS / 2)) * 2;
    float e0 = 0.f, e1 = 0.f;
    if (rr < np) {
      const float mxr = rmax[rr];
      if (c0 < NC)     e0 = __expf(tile[rr * NC + c0] - mxr);
      if (c0 + 1 < NC) e1 = __expf(tile[rr * NC + c0 + 1] - mxr);
    }
    et[rr * ECOLS + c0]     = (_Float16)e0;
    et[rr * ECOLS + c0 + 1] = (_Float16)e1;
  }
  __syncthreads();

  // ---- row sums via v_wmma_f32_16x16x32_f16, B = ones ----
  // ISA 16-bit A 16x32 layout: lane half hi holds K in [hi*8,hi*8+7] and
  // [16+hi*8, 16+hi*8+7] -> two aligned 16B LDS loads per K-step.
  v16h bones;
#pragma unroll
  for (int j = 0; j < 16; ++j) bones[j] = (_Float16)1.0f;
  v8f acc = {};
#pragma unroll
  for (int kb = 0; kb < ECOLS; kb += 32) {
    const v8h* ep = (const v8h*)&et[r * ECOLS + kb + hi * 8];
    const v8h alo = ep[0];   // K = kb + hi*8 + 0..7
    const v8h ahi = ep[2];   // K = kb + 16 + hi*8 + 0..7
    const v16h a = __builtin_shufflevector(alo, ahi, 0, 1, 2, 3, 4, 5, 6, 7,
                                           8, 9, 10, 11, 12, 13, 14, 15);
    acc = __builtin_amdgcn_wmma_f32_16x16x32_f16(false, a, false, bones,
                                                 (short)0, acc, false, false);
  }
  // D layout: VGPR v = row M=v (lanes 0-15) / M=v+8 (lanes 16-31)
  if (m == 0) {
#pragma unroll
    for (int v2 = 0; v2 < 8; ++v2) wsum[w * 16 + hi * 8 + v2] = acc[v2];
  }
  __syncthreads();
  const float rowsum = wsum[w * 16 + m];

  // ---- CE, mine array, positive bookkeeping ----
  float posce = 0.f;
  int   cnt   = 0;
  if (hi == 0 && r < np) {
    const int   p   = p0 + r;
    const int   tgt = conf_t[(size_t)b * NP + p];
    const float ce  = mx + __logf(rowsum) - tile[r * NC + tgt];
    mine[(size_t)b * NP + p] = (tgt > 0) ? 0.f : ce;
    if (tgt > 0) { posce = ce; cnt = 1; }
  }
  posce = wave_sum_f(posce);
  cnt   = wave_sum_i(cnt);
  if (lane == 0) {
    if (posce != 0.f) atomicAdd(&fws[2], posce);
    if (cnt) { atomicAdd(&iws[8 + b], cnt); atomicAdd(&iws[4], cnt); }
  }
}

__global__ void mine_kernel(const float* __restrict__ mine,
                            const int* __restrict__ iws, float* fws) {
  __shared__ float s[SORT_N];  // 64 KB, fits WGP LDS
  const int b   = blockIdx.x;
  const int tid = threadIdx.x;
  const int nt  = blockDim.x;
  for (int i = tid; i < SORT_N; i += nt)
    s[i] = (i < NP) ? mine[(size_t)b * NP + i] : 0.f;
  __syncthreads();
  for (int k = 2; k <= SORT_N; k <<= 1) {
    for (int j = k >> 1; j > 0; j >>= 1) {
      for (int i = tid; i < SORT_N; i += nt) {
        const int ij = i ^ j;
        if (ij > i) {
          const float a = s[i], c = s[ij];
          const bool desc = ((i & k) == 0);
          if (desc ? (a < c) : (a > c)) { s[i] = c; s[ij] = a; }
        }
      }
      __syncthreads();
    }
  }
  const int npos = iws[8 + b];
  const int nneg = min(NEG_POS_RATIO * npos, NP - 1);
  float part = 0.f;
  for (int i = tid; i < nneg; i += nt) part += s[i];
  __syncthreads();
  s[tid] = part;
  __syncthreads();
  for (int off = nt >> 1; off > 0; off >>= 1) {
    if (tid < off) s[tid] += s[tid + off];
    __syncthreads();
  }
  if (tid == 0) atomicAdd(&fws[2], s[0]);
}

__global__ void finalize_kernel(const float* fws, const int* iws, float* out) {
  const float N = (float)iws[4];
  out[0] = fws[0] / N;   // loss_l / N
  out[1] = fws[2] / N;   // loss_c / N
  out[2] = fws[1] / N;   // loss_four_corners / N
}

extern "C" void kernel_launch(void* const* d_in, const int* in_sizes, int n_in,
                              void* d_out, int out_size, void* d_ws, size_t ws_size,
                              hipStream_t stream) {
  (void)in_sizes; (void)n_in; (void)out_size; (void)ws_size;
  const float* loc_d  = (const float*)d_in[0];
  const float* conf_d = (const float*)d_in[1];
  const float* fc_d   = (const float*)d_in[2];
  const float* loc_t  = (const float*)d_in[3];
  const float* fc_t   = (const float*)d_in[4];
  const int*   conf_t = (const int*)d_in[5];
  float* out  = (float*)d_out;
  float* fws  = (float*)d_ws;
  int*   iws  = (int*)d_ws;
  float* mine = fws + 256;

  zero_ws_kernel<<<1, 128, 0, stream>>>(iws);
  smoothl1_kernel<<<(NB * NP + 255) / 256, 256, 0, stream>>>(loc_d, loc_t, fc_d,
                                                             fc_t, conf_t, fws);
  dim3 ceg((NP + TILE_P - 1) / TILE_P, NB);
  ce_kernel<<<ceg, NTHR, 0, stream>>>(conf_d, conf_t, fws, iws, mine);
  mine_kernel<<<NB, 1024, 0, stream>>>(mine, iws, fws);
  finalize_kernel<<<1, 1, 0, stream>>>(fws, iws, out);
}
